// TSPEmbGNN_54125177865091
// MI455X (gfx1250) — matrix operand order repaired and checked by reading, not compiled, not run on hardware
//
#include <hip/hip_runtime.h>
#include <math.h>

#define D 32
#define GDEPTH 12
#define EPS 1e-5f

typedef __attribute__((ext_vector_type(16))) __bf16 v16bf;
typedef __attribute__((ext_vector_type(8)))  float  v8f;

__device__ __forceinline__ float siluf(float x) { return x / (1.f + __expf(-x)); }
__device__ __forceinline__ float sigmf(float x) { return 1.f / (1.f + __expf(-x)); }

// ---------------- small elementwise kernels ----------------

__global__ void k_zero(float* __restrict__ p, int n) {
    int i = blockIdx.x * blockDim.x + threadIdx.x;
    if (i < n) p[i] = 0.f;
}

// Pack f32 weight matrices [nmat][K=32][C=32] (row-major, y = x@W) into the
// exact WMMA B-operand layout, bf16: P[m][cn][lane][j] where
//   lg = lane>>4, lr = lane&15, c = cn*16+lr, k = (j<8) ? lg*8 + j : 16 + lg*8 + (j-8).
__global__ void k_pack_weights(const float* __restrict__ W, __bf16* __restrict__ P, int nmat) {
    int i = blockIdx.x * blockDim.x + threadIdx.x;
    if (i >= nmat * 1024) return;
    int m = i >> 10;
    int p = i & 1023;
    int cn = p >> 9, lane = (p >> 4) & 31, j = p & 15;
    int lg = lane >> 4, lr = lane & 15;
    int c = cn * 16 + lr;
    int k = (j < 8) ? lg * 8 + j : 16 + lg * 8 + (j - 8);
    P[i] = (__bf16)W[m * 1024 + k * D + c];
}

// h = silu(x @ Wv0 + bv0), x:[N,2], Wv0:[2,32]
__global__ void k_embed_nodes(const float* __restrict__ x, const float* __restrict__ Wv0,
                              const float* __restrict__ bv0, float* __restrict__ h, int n) {
    int i = blockIdx.x * blockDim.x + threadIdx.x;
    if (i >= n * D) return;
    int node = i >> 5, c = i & 31;
    float v = x[node * 2 + 0] * Wv0[c] + x[node * 2 + 1] * Wv0[D + c] + bv0[c];
    h[i] = siluf(v);
}

// w = silu(edge_attr @ We0 + be0), edge_attr:[E,1], We0:[1,32]
__global__ void k_embed_edges(const float* __restrict__ ea, const float* __restrict__ We0,
                              const float* __restrict__ be0, float* __restrict__ w, int E) {
    int i = blockIdx.x * blockDim.x + threadIdx.x;
    if (i >= E * D) return;
    int e = i >> 5, c = i & 31;
    w[i] = siluf(ea[e] * We0[c] + be0[c]);
}

__global__ void k_deg_count(const int* __restrict__ src, float* __restrict__ deg, int E) {
    int i = blockIdx.x * blockDim.x + threadIdx.x;
    if (i < E) atomicAdd(&deg[src[i]], 1.f);
}

__global__ void k_deg_inv(float* __restrict__ deg, int n) {
    int i = blockIdx.x * blockDim.x + threadIdx.x;
    if (i < n) deg[i] = 1.f / fmaxf(deg[i], 1.f);
}

// ---------------- WMMA node GEMM: x1..x4 = h @ Wv{1..4} + b ----------------
// One wave = one 16-row tile; A built from f32 h per ISA 16-bit A layout
// (lanes 0-15: K 0-7 / 16-23; lanes 16-31: K 8-15 / 24-31), B prepacked bf16.
__global__ void k_node_gemm4(const float* __restrict__ h,
                             const __bf16* __restrict__ P1, const float* __restrict__ b1,
                             const __bf16* __restrict__ P2, const float* __restrict__ b2,
                             const __bf16* __restrict__ P3, const float* __restrict__ b3,
                             const __bf16* __restrict__ P4, const float* __restrict__ b4,
                             float* __restrict__ o1, float* __restrict__ o2,
                             float* __restrict__ o3, float* __restrict__ o4, int n) {
    int wave = threadIdx.x >> 5, lane = threadIdx.x & 31;
    int lg = lane >> 4, lr = lane & 15;
    int rowBase = (blockIdx.x * (blockDim.x >> 5) + wave) * 16;
    if (rowBase >= n) return;      // wave-uniform
    bool full = (rowBase + 16 <= n);  // wave-uniform fast path

    int r = rowBase + lr;
    if (r >= n) r = n - 1;
    const float* hp = h + (size_t)r * D;
    int k0 = lg ? 8 : 0, k1 = lg ? 24 : 16;
    v16bf a;
#pragma unroll
    for (int j = 0; j < 8; ++j) { a[j] = (__bf16)hp[k0 + j]; a[8 + j] = (__bf16)hp[k1 + j]; }

    const __bf16* Ps[4] = {P1, P2, P3, P4};
    const float* bs[4] = {b1, b2, b3, b4};
    float* os[4] = {o1, o2, o3, o4};
#pragma unroll
    for (int m = 0; m < 4; ++m) {
#pragma unroll
        for (int cn = 0; cn < 2; ++cn) {
            v16bf b = ((const v16bf*)Ps[m])[cn * 32 + lane];
            v8f acc = {};
            acc = __builtin_amdgcn_wmma_f32_16x16x32_bf16(false, a, false, b, (short)0, acc,
                                                          false, false);
            int c = cn * 16 + lr;
            float bb = bs[m][c];
            float* op = os[m] + (size_t)(rowBase + 8 * lg) * D + c;
            if (full) {
#pragma unroll
                for (int v = 0; v < 8; ++v) op[(size_t)v * D] = acc[v] + bb;
            } else {
#pragma unroll
                for (int v = 0; v < 8; ++v)
                    if (rowBase + 8 * lg + v < n) op[(size_t)v * D] = acc[v] + bb;
            }
        }
    }
}

// ---------------- fused edge pass ----------------
// w1 = w0 @ We + be (WMMA); msg = sigmoid(w0)*x2[dst] scatter-added into agg;
// tmp = w1 + x3[src] + x4[dst]; BN stats (sum, sumsq) reduced via LDS.
__global__ void k_edge_pass(const float* __restrict__ w, const __bf16* __restrict__ Pe,
                            const float* __restrict__ bei, const float* __restrict__ x2,
                            const float* __restrict__ x3, const float* __restrict__ x4,
                            const int* __restrict__ src, const int* __restrict__ dst,
                            float* __restrict__ agg, float* __restrict__ tmpe,
                            float* __restrict__ esum, float* __restrict__ esq, int E) {
    __shared__ float ssum[D], ssq[D];
    int tid = threadIdx.x;
    if (tid < D) { ssum[tid] = 0.f; ssq[tid] = 0.f; }
    __syncthreads();

    int wave = tid >> 5, lane = tid & 31;
    int lg = lane >> 4, lr = lane & 15;
    int rowBase = (blockIdx.x * (blockDim.x >> 5) + wave) * 16;
    bool active = rowBase < E;       // wave-uniform; no early return (barrier below)
    bool full = (rowBase + 16 <= E); // wave-uniform

    if (active) {
        int er = rowBase + lr;
        if (er >= E) er = E - 1;
        const float* wp = w + (size_t)er * D;
        int k0 = lg ? 8 : 0, k1 = lg ? 24 : 16;
        v16bf a;
#pragma unroll
        for (int j = 0; j < 8; ++j) { a[j] = (__bf16)wp[k0 + j]; a[8 + j] = (__bf16)wp[k1 + j]; }

        // edge indices for this lane-group's 8 rows (shared across both halves)
        int eBase = rowBase + 8 * lg;
        int sArr[8], dArr[8];
#pragma unroll
        for (int v = 0; v < 8; ++v) {
            int e = eBase + v;
            if (e >= E) e = E - 1;
            sArr[v] = src[e];
            dArr[v] = dst[e];
        }

#pragma unroll
        for (int cn = 0; cn < 2; ++cn) {
            v16bf b = ((const v16bf*)Pe)[cn * 32 + lane];
            v8f acc = {};
            acc = __builtin_amdgcn_wmma_f32_16x16x32_bf16(false, a, false, b, (short)0, acc,
                                                          false, false);
            int c = cn * 16 + lr;
            float bb = bei[c];
            float lsum = 0.f, lsq = 0.f;
            if (full) {
#pragma unroll
                for (int v = 0; v < 8; ++v) {
                    int e = eBase + v;
                    int s = sArr[v], dd = dArr[v];
                    float w0v = w[(size_t)e * D + c];
                    float msg = sigmf(w0v) * x2[(size_t)dd * D + c];
                    atomicAdd(&agg[(size_t)s * D + c], msg);
                    float t = acc[v] + bb + x3[(size_t)s * D + c] + x4[(size_t)dd * D + c];
                    tmpe[(size_t)e * D + c] = t;
                    lsum += t;
                    lsq += t * t;
                }
            } else {
#pragma unroll
                for (int v = 0; v < 8; ++v) {
                    int e = eBase + v;
                    if (e < E) {
                        int s = sArr[v], dd = dArr[v];
                        float w0v = w[(size_t)e * D + c];
                        float msg = sigmf(w0v) * x2[(size_t)dd * D + c];
                        atomicAdd(&agg[(size_t)s * D + c], msg);
                        float t = acc[v] + bb + x3[(size_t)s * D + c] + x4[(size_t)dd * D + c];
                        tmpe[(size_t)e * D + c] = t;
                        lsum += t;
                        lsq += t * t;
                    }
                }
            }
            atomicAdd(&ssum[c], lsum);
            atomicAdd(&ssq[c], lsq);
        }
    }
    __syncthreads();
    if (tid < D) { atomicAdd(&esum[tid], ssum[tid]); atomicAdd(&esq[tid], ssq[tid]); }
}

// t = x1 + agg/deg (stored in-place into x1); accumulate node BN stats.
__global__ void k_node_stats(float* __restrict__ x1, const float* __restrict__ agg,
                             const float* __restrict__ dinv, float* __restrict__ vsum,
                             float* __restrict__ vsq, int n) {
    __shared__ float ssum[D], ssq[D];
    int tid = threadIdx.x;
    if (tid < D) { ssum[tid] = 0.f; ssq[tid] = 0.f; }
    __syncthreads();
    int i = blockIdx.x * blockDim.x + tid;
    int c = tid & 31;
    float t = 0.f, t2 = 0.f;
    if (i < n * D) {
        float v = x1[i] + agg[i] * dinv[i >> 5];
        x1[i] = v;
        t = v;
        t2 = v * v;
    }
    atomicAdd(&ssum[c], t);
    atomicAdd(&ssq[c], t2);
    __syncthreads();
    if (tid < D) { atomicAdd(&vsum[tid], ssum[tid]); atomicAdd(&vsq[tid], ssq[tid]); }
}

// stats out: [mean_v, rs_v, mean_e, rs_e] each [32]
__global__ void k_finalize(const float* __restrict__ acc4, float* __restrict__ stats, int n,
                           int E) {
    int c = threadIdx.x;
    if (c < D) {
        float mv = acc4[c] / (float)n;
        float varv = acc4[D + c] / (float)n - mv * mv;
        stats[c] = mv;
        stats[D + c] = rsqrtf(fmaxf(varv, 0.f) + EPS);
        float me = acc4[2 * D + c] / (float)E;
        float vare = acc4[3 * D + c] / (float)E - me * me;
        stats[2 * D + c] = me;
        stats[3 * D + c] = rsqrtf(fmaxf(vare, 0.f) + EPS);
    }
}

__global__ void k_node_update(float* __restrict__ h, const float* __restrict__ t,
                              const float* __restrict__ stats, const float* __restrict__ g,
                              const float* __restrict__ b, int n) {
    int i = blockIdx.x * blockDim.x + threadIdx.x;
    if (i >= n * D) return;
    int c = i & 31;
    float v = (t[i] - stats[c]) * stats[D + c] * g[c] + b[c];
    h[i] += siluf(v);
}

__global__ void k_edge_update(const float* __restrict__ w, const float* __restrict__ tmpe,
                              const float* __restrict__ stats, const float* __restrict__ g,
                              const float* __restrict__ b, float* __restrict__ out, int E) {
    int i = blockIdx.x * blockDim.x + threadIdx.x;
    if (i >= E * D) return;
    // warm L2/L0 ahead on the two streamed inputs (global_prefetch_b8)
    __builtin_prefetch(tmpe + i + 16384, 0, 0);
    __builtin_prefetch(w + i + 16384, 0, 0);
    int c = i & 31;
    float v = (tmpe[i] - stats[2 * D + c]) * stats[3 * D + c] * g[c] + b[c];
    out[i] = w[i] + siluf(v);
}

// ---------------- launcher ----------------

extern "C" void kernel_launch(void* const* d_in, const int* in_sizes, int n_in, void* d_out,
                              int out_size, void* d_ws, size_t ws_size, hipStream_t stream) {
    const float* x    = (const float*)d_in[0];
    const int*   ei   = (const int*)d_in[1];
    const float* ea   = (const float*)d_in[2];
    const float* Wv0  = (const float*)d_in[3];
    const float* bv0  = (const float*)d_in[4];
    const float* Wv1  = (const float*)d_in[5];
    const float* bv1  = (const float*)d_in[6];
    const float* Wv2  = (const float*)d_in[7];
    const float* bv2  = (const float*)d_in[8];
    const float* Wv3  = (const float*)d_in[9];
    const float* bv3  = (const float*)d_in[10];
    const float* Wv4  = (const float*)d_in[11];
    const float* bv4  = (const float*)d_in[12];
    const float* gv   = (const float*)d_in[13];
    const float* betv = (const float*)d_in[14];
    const float* We0  = (const float*)d_in[15];
    const float* be0  = (const float*)d_in[16];
    const float* We   = (const float*)d_in[17];
    const float* be   = (const float*)d_in[18];
    const float* ge   = (const float*)d_in[19];
    const float* bete = (const float*)d_in[20];

    const int N = in_sizes[0] / 2;
    const int E = in_sizes[2];

    float* ws = (float*)d_ws;
    size_t off = 0;
    auto alloc = [&](size_t nel) {
        float* p = ws + off;
        off += (nel + (size_t)255) & ~(size_t)255;
        return p;
    };
    float* h     = alloc((size_t)N * D);
    float* x1    = alloc((size_t)N * D);
    float* x2    = alloc((size_t)N * D);
    float* x3    = alloc((size_t)N * D);
    float* x4    = alloc((size_t)N * D);
    float* agg   = alloc((size_t)N * D);
    float* dinv  = alloc((size_t)N);
    float* acc4  = alloc(4 * D);   // vsum, vsq, esum, esq
    float* stats = alloc(4 * D);   // mean_v, rs_v, mean_e, rs_e
    __bf16* pW   = (__bf16*)alloc((size_t)5 * GDEPTH * 1024 / 2);  // packed weights (bf16)
    float* w     = alloc((size_t)E * D);
    float* tmpe  = alloc((size_t)E * D);

    __bf16* pWv1 = pW + 0 * GDEPTH * 1024;
    __bf16* pWv2 = pW + 1 * GDEPTH * 1024;
    __bf16* pWv3 = pW + 2 * GDEPTH * 1024;
    __bf16* pWv4 = pW + 3 * GDEPTH * 1024;
    __bf16* pWe  = pW + 4 * GDEPTH * 1024;

    const int* srcIdx = ei;
    const int* dstIdx = ei + E;

    const int BT = 256;
    int gN = (N * D + BT - 1) / BT;
    int gE = (E * D + BT - 1) / BT;
    int gTileN = (N + 127) / 128;  // 8 waves x 16 rows per block
    int gTileE = (E + 127) / 128;
    int gPack = (GDEPTH * 1024 + BT - 1) / BT;

    // ---- setup ----
    k_pack_weights<<<gPack, BT, 0, stream>>>(Wv1, pWv1, GDEPTH);
    k_pack_weights<<<gPack, BT, 0, stream>>>(Wv2, pWv2, GDEPTH);
    k_pack_weights<<<gPack, BT, 0, stream>>>(Wv3, pWv3, GDEPTH);
    k_pack_weights<<<gPack, BT, 0, stream>>>(Wv4, pWv4, GDEPTH);
    k_pack_weights<<<gPack, BT, 0, stream>>>(We, pWe, GDEPTH);
    k_embed_nodes<<<gN, BT, 0, stream>>>(x, Wv0, bv0, h, N);
    k_embed_edges<<<gE, BT, 0, stream>>>(ea, We0, be0, w, E);
    k_zero<<<(N + BT - 1) / BT, BT, 0, stream>>>(dinv, N);
    k_deg_count<<<(E + BT - 1) / BT, BT, 0, stream>>>(srcIdx, dinv, E);
    k_deg_inv<<<(N + BT - 1) / BT, BT, 0, stream>>>(dinv, N);

    // ---- layers ----
    for (int l = 0; l < GDEPTH; ++l) {
        k_zero<<<gN, BT, 0, stream>>>(agg, N * D);
        k_zero<<<1, 4 * D, 0, stream>>>(acc4, 4 * D);
        k_node_gemm4<<<gTileN, BT, 0, stream>>>(h, pWv1 + l * 1024, bv1 + l * D,
                                                pWv2 + l * 1024, bv2 + l * D,
                                                pWv3 + l * 1024, bv3 + l * D,
                                                pWv4 + l * 1024, bv4 + l * D, x1, x2, x3, x4, N);
        k_edge_pass<<<gTileE, BT, 0, stream>>>(w, pWe + l * 1024, be + l * D, x2, x3, x4, srcIdx,
                                               dstIdx, agg, tmpe, acc4 + 2 * D, acc4 + 3 * D, E);
        k_node_stats<<<gN, BT, 0, stream>>>(x1, agg, dinv, acc4, acc4 + D, N);
        k_finalize<<<1, D, 0, stream>>>(acc4, stats, N, E);
        k_node_update<<<gN, BT, 0, stream>>>(h, x1, stats, gv + l * D, betv + l * D, N);
        float* dest = (l == GDEPTH - 1) ? (float*)d_out : w;
        k_edge_update<<<gE, BT, 0, stream>>>(w, tmpe, stats, ge + l * D, bete + l * D, dest, E);
    }
}
